// MoELayer_1322849927962
// MI455X (gfx1250) — matrix operand order repaired and checked by reading, not compiled, not run on hardware
//
#include <hip/hip_runtime.h>
#include <hip/hip_bf16.h>
#include <math.h>

#define NUM_EXPERTS 8
#define MODEL_DIM   1024
#define HIDDEN      2048
#define TOKENS      8192

typedef __attribute__((ext_vector_type(16))) __bf16 v16bf;
typedef __attribute__((ext_vector_type(8)))  float  v8f;

union ABOp { v16bf v; uint4 q[2]; };

// ---------------------------------------------------------------------------
// init: zero expert counts
// ---------------------------------------------------------------------------
__global__ void moe_init_kernel(int* counts) {
    if (threadIdx.x < NUM_EXPERTS) counts[threadIdx.x] = 0;
}

// ---------------------------------------------------------------------------
// transpose + f32->bf16: src [E][R][C] -> dst [E][C][R]
// ---------------------------------------------------------------------------
__global__ void transpose_bf16_kernel(const float* __restrict__ src,
                                      __bf16* __restrict__ dst, int R, int C) {
    __shared__ float tile[32][33];
    const int e  = blockIdx.z;
    const float* s = src + (size_t)e * R * C;
    __bf16*      d = dst + (size_t)e * R * C;
    const int c0 = blockIdx.x * 32;
    const int r0 = blockIdx.y * 32;
    const int tx = threadIdx.x;   // 0..31
    const int ty = threadIdx.y;   // 0..7
    #pragma unroll
    for (int k = 0; k < 4; ++k) {
        int r = r0 + ty + 8 * k;
        tile[ty + 8 * k][tx] = s[(size_t)r * C + (c0 + tx)];
    }
    __syncthreads();
    #pragma unroll
    for (int k = 0; k < 4; ++k) {
        int c = c0 + ty + 8 * k;   // dst row
        int r = r0 + tx;           // dst col
        d[(size_t)c * R + r] = (__bf16)tile[tx][ty + 8 * k];
    }
}

// ---------------------------------------------------------------------------
// gate: one wave32 per token -> top-1 expert, count per expert
// ---------------------------------------------------------------------------
__global__ void gate_kernel(const float* __restrict__ x,
                            const float* __restrict__ gw,
                            const float* __restrict__ gb,
                            int* __restrict__ sel, int* __restrict__ counts) {
    const int wave = threadIdx.x >> 5;
    const int lane = threadIdx.x & 31;
    const int t = blockIdx.x * 8 + wave;
    const float* xr = x + (size_t)t * MODEL_DIM;
    float acc[NUM_EXPERTS];
    #pragma unroll
    for (int e = 0; e < NUM_EXPERTS; ++e) acc[e] = 0.0f;
    for (int d = lane; d < MODEL_DIM; d += 32) {
        float xv = xr[d];
        const float* g = gw + d * NUM_EXPERTS;
        #pragma unroll
        for (int e = 0; e < NUM_EXPERTS; ++e) acc[e] += xv * g[e];
    }
    #pragma unroll
    for (int e = 0; e < NUM_EXPERTS; ++e)
        for (int off = 16; off > 0; off >>= 1)
            acc[e] += __shfl_xor(acc[e], off, 32);
    if (lane == 0) {
        int best = 0; float bv = acc[0] + gb[0];
        #pragma unroll
        for (int e = 1; e < NUM_EXPERTS; ++e) {
            float v = acc[e] + gb[e];
            if (v > bv) { bv = v; best = e; }   // strict > : lowest index wins
        }
        sel[t] = best;
        atomicAdd(&counts[best], 1);
    }
}

// ---------------------------------------------------------------------------
// scan: exclusive prefix over 8 counts; also writes l_aux = 0 (tuple output)
// ---------------------------------------------------------------------------
__global__ void scan_kernel(const int* __restrict__ counts, int* __restrict__ offsets,
                            int* __restrict__ cursors, float* __restrict__ laux) {
    if (threadIdx.x == 0) {
        int acc = 0;
        for (int e = 0; e < NUM_EXPERTS; ++e) {
            offsets[e] = acc; cursors[e] = acc; acc += counts[e];
        }
        offsets[NUM_EXPERTS] = acc;
        *laux = 0.0f;
    }
}

// ---------------------------------------------------------------------------
// scatter: compact token ids per expert
// ---------------------------------------------------------------------------
__global__ void scatter_kernel(const int* __restrict__ sel, int* __restrict__ cursors,
                               int* __restrict__ perm) {
    int t = blockIdx.x * blockDim.x + threadIdx.x;
    int e = sel[t];
    int pos = atomicAdd(&cursors[e], 1);
    perm[pos] = t;
}

// ---------------------------------------------------------------------------
// fused expert FFN: 32 gathered tokens per workgroup, bf16 WMMA both passes.
// Each wave owns the FULL 32-row M stripe for its N tiles, so every B tile
// loaded from L2 feeds TWO v_wmma (M tile 0 and 1) -> half the weight traffic.
// ---------------------------------------------------------------------------
#define XS_STRIDE 1032   // 1024 + 8 bf16 pad (16B) -> low-conflict b128 LDS reads
#define HS_STRIDE 2056   // 2048 + 8

__global__ __launch_bounds__(256, 1)
void moe_ffn_kernel(const float* __restrict__ x,
                    const __bf16* __restrict__ w1t,   // [E][H][D] bf16 (pre-transposed)
                    const float* __restrict__ b1,     // [E][H]
                    const __bf16* __restrict__ w2t,   // [E][D][H] bf16 (pre-transposed)
                    const float* __restrict__ b2,     // [E][D]
                    const int* __restrict__ offsets,
                    const int* __restrict__ perm,
                    float* __restrict__ out) {
    __shared__ __bf16 xs[32 * XS_STRIDE];   // 66 KB
    __shared__ __bf16 hs[32 * HS_STRIDE];   // 128.5 KB
    __shared__ int    toks[32];

    const int e    = blockIdx.y;
    const int base = offsets[e];
    const int cnt  = offsets[e + 1] - base;
    const int t0   = blockIdx.x * 32;
    if (t0 >= cnt) return;

    const int tid = threadIdx.x;
    if (tid < 32) toks[tid] = (t0 + tid < cnt) ? perm[base + t0 + tid] : -1;
    __syncthreads();

    // gather x tile: fp32 global -> bf16 LDS (coalesced)
    for (int idx = tid; idx < 32 * MODEL_DIM; idx += 256) {
        int r = idx >> 10, c = idx & (MODEL_DIM - 1);
        int tok = toks[r];
        float v = (tok >= 0) ? x[(size_t)tok * MODEL_DIM + c] : 0.0f;
        xs[r * XS_STRIDE + c] = (__bf16)v;
    }
    __syncthreads();

    const int wave  = tid >> 5;           // 0..7 : N-tile phase
    const int lane  = tid & 31;
    const int lhalf = (lane < 16) ? 0 : 1;
    const int ln    = lane & 15;
    const int kbA   = lhalf * 8;          // A: halves 0-7 -> K c0..c0+7, 8-15 -> c0+16..c0+23
    const int kbB   = lhalf * 8;          // B: same K order as A (halves 0-7 -> kb..kb+7, 8-15 -> kb+16..kb+23)

    const __bf16* xrow0 = xs + ln * XS_STRIDE;          // M tile 0 rows 0..15
    const __bf16* xrow1 = xs + (ln + 16) * XS_STRIDE;   // M tile 1 rows 16..31
    const __bf16* w1e  = w1t + (size_t)e * HIDDEN * MODEL_DIM;
    const float*  b1e  = b1 + e * HIDDEN;
    const v8f vzero = {0.f, 0.f, 0.f, 0.f, 0.f, 0.f, 0.f, 0.f};

    // ---- Pass 1: h = gelu(x @ w1 + b1); 128 N-tiles over H, 16/wave, groups of 4
    for (int g = 0; g < 4; ++g) {
        v8f acc[2][4];
        int ncol[4];
        #pragma unroll
        for (int u = 0; u < 4; ++u) {
            ncol[u] = (wave + 8 * (4 * g + u)) * 16 + ln;
            acc[0][u] = vzero; acc[1][u] = vzero;
        }
        for (int kt = 0; kt < MODEL_DIM / 32; ++kt) {
            ABOp a0, a1;
            const int c0 = kt * 32 + kbA;
            a0.q[0] = *(const uint4*)(xrow0 + c0);
            a0.q[1] = *(const uint4*)(xrow0 + c0 + 16);
            a1.q[0] = *(const uint4*)(xrow1 + c0);
            a1.q[1] = *(const uint4*)(xrow1 + c0 + 16);
            const int kb = kt * 32 + kbB;
            #pragma unroll
            for (int u = 0; u < 4; ++u) {
                ABOp b;
                const __bf16* bp = w1e + (size_t)ncol[u] * MODEL_DIM + kb;
                b.q[0] = ((const uint4*)bp)[0];
                b.q[1] = ((const uint4*)bp)[2];
                acc[0][u] = __builtin_amdgcn_wmma_f32_16x16x32_bf16(
                    false, a0.v, false, b.v, (short)0, acc[0][u], false, false);
                acc[1][u] = __builtin_amdgcn_wmma_f32_16x16x32_bf16(
                    false, a1.v, false, b.v, (short)0, acc[1][u], false, false);
            }
        }
        #pragma unroll
        for (int u = 0; u < 4; ++u) {
            const float bias = b1e[ncol[u]];
            #pragma unroll
            for (int mt = 0; mt < 2; ++mt) {
                #pragma unroll
                for (int r = 0; r < 8; ++r) {
                    float v = acc[mt][u][r] + bias;
                    float gel = 0.5f * v * (1.0f + erff(v * 0.70710678118f)); // exact erf GELU
                    int m = r + 8 * lhalf + 16 * mt;
                    hs[m * HS_STRIDE + ncol[u]] = (__bf16)gel;
                }
            }
        }
    }
    __syncthreads();

    const __bf16* w2e  = w2t + (size_t)e * MODEL_DIM * HIDDEN;
    const float*  b2e  = b2 + e * MODEL_DIM;
    const __bf16* hrow0 = hs + ln * HS_STRIDE;
    const __bf16* hrow1 = hs + (ln + 16) * HS_STRIDE;

    // ---- Pass 2: y = h @ w2 + b2; 64 N-tiles over D, 8/wave, groups of 4
    for (int g = 0; g < 2; ++g) {
        v8f acc[2][4];
        int ncol[4];
        #pragma unroll
        for (int u = 0; u < 4; ++u) {
            ncol[u] = (wave + 8 * (4 * g + u)) * 16 + ln;
            acc[0][u] = vzero; acc[1][u] = vzero;
        }
        for (int kt = 0; kt < HIDDEN / 32; ++kt) {
            ABOp a0, a1;
            const int c0 = kt * 32 + kbA;
            a0.q[0] = *(const uint4*)(hrow0 + c0);
            a0.q[1] = *(const uint4*)(hrow0 + c0 + 16);
            a1.q[0] = *(const uint4*)(hrow1 + c0);
            a1.q[1] = *(const uint4*)(hrow1 + c0 + 16);
            const int kb = kt * 32 + kbB;
            #pragma unroll
            for (int u = 0; u < 4; ++u) {
                ABOp b;
                const __bf16* bp = w2e + (size_t)ncol[u] * HIDDEN + kb;
                b.q[0] = ((const uint4*)bp)[0];
                b.q[1] = ((const uint4*)bp)[2];
                acc[0][u] = __builtin_amdgcn_wmma_f32_16x16x32_bf16(
                    false, a0.v, false, b.v, (short)0, acc[0][u], false, false);
                acc[1][u] = __builtin_amdgcn_wmma_f32_16x16x32_bf16(
                    false, a1.v, false, b.v, (short)0, acc[1][u], false, false);
            }
        }
        #pragma unroll
        for (int u = 0; u < 4; ++u) {
            const float bias = b2e[ncol[u]];
            #pragma unroll
            for (int mt = 0; mt < 2; ++mt) {
                #pragma unroll
                for (int r = 0; r < 8; ++r) {
                    int m = r + 8 * lhalf + 16 * mt;
                    int tok = toks[m];
                    if (tok >= 0)
                        out[(size_t)tok * MODEL_DIM + ncol[u]] = acc[mt][u][r] + bias;
                }
            }
        }
    }
}

// ---------------------------------------------------------------------------
extern "C" void kernel_launch(void* const* d_in, const int* in_sizes, int n_in,
                              void* d_out, int out_size, void* d_ws, size_t ws_size,
                              hipStream_t stream) {
    (void)in_sizes; (void)n_in; (void)out_size; (void)ws_size;
    const float* x      = (const float*)d_in[0];
    const float* gate_w = (const float*)d_in[1];
    const float* gate_b = (const float*)d_in[2];
    const float* w1     = (const float*)d_in[3];
    const float* b1     = (const float*)d_in[4];
    const float* w2     = (const float*)d_in[5];
    const float* b2     = (const float*)d_in[6];
    float* out = (float*)d_out;

    char* ws = (char*)d_ws;
    __bf16* w1t   = (__bf16*)ws;                                  // 32 MB
    __bf16* w2t   = (__bf16*)(ws + ((size_t)32 << 20));           // 32 MB
    int* sel      = (int*)(ws + ((size_t)64 << 20));              // 8192 ints
    int* counts   = sel + TOKENS;                                 // 8
    int* offsets  = counts + NUM_EXPERTS;                         // 9
    int* cursors  = offsets + NUM_EXPERTS + 1;                    // 8
    int* perm     = cursors + NUM_EXPERTS;                        // 8192

    moe_init_kernel<<<1, 32, 0, stream>>>(counts);
    // w1 [E][D][H] -> w1t [E][H][D] bf16
    transpose_bf16_kernel<<<dim3(HIDDEN / 32, MODEL_DIM / 32, NUM_EXPERTS),
                            dim3(32, 8), 0, stream>>>(w1, w1t, MODEL_DIM, HIDDEN);
    // w2 [E][H][D2] -> w2t [E][D2][H] bf16
    transpose_bf16_kernel<<<dim3(MODEL_DIM / 32, HIDDEN / 32, NUM_EXPERTS),
                            dim3(32, 8), 0, stream>>>(w2, w2t, HIDDEN, MODEL_DIM);
    gate_kernel<<<TOKENS / 8, 256, 0, stream>>>(x, gate_w, gate_b, sel, counts);
    scan_kernel<<<1, 32, 0, stream>>>(counts, offsets, cursors,
                                      out + (size_t)TOKENS * MODEL_DIM);
    scatter_kernel<<<TOKENS / 256, 256, 0, stream>>>(sel, cursors, perm);
    moe_ffn_kernel<<<dim3(TOKENS / 32, NUM_EXPERTS), 256, 0, stream>>>(
        x, w1t, b1, w2t, b2, offsets, perm, out);
}
